// First_52699248722071
// MI455X (gfx1250) — matrix operand order repaired and checked by reading, not compiled, not run on hardware
//
#include <hip/hip_runtime.h>
#include <math.h>

typedef __attribute__((ext_vector_type(16))) _Float16 v16h;
typedef __attribute__((ext_vector_type(8)))  _Float16 v8h;
typedef __attribute__((ext_vector_type(2)))  _Float16 v2h;
typedef __attribute__((ext_vector_type(8)))  float    v8f;
typedef __attribute__((ext_vector_type(4)))  int      v4i;

#define NWAVES 8
#define EPSF   1e-12f

// ---------------------------------------------------------------------------
// zero the pooled accumulator (workspace is poisoned, must re-init every call)
// ---------------------------------------------------------------------------
__global__ void zero_kernel(float* __restrict__ p, int n) {
  int i = blockIdx.x * blockDim.x + threadIdx.x;
  int stride = gridDim.x * blockDim.x;
  for (; i < n; i += stride) p[i] = 0.0f;
}

// ---------------------------------------------------------------------------
// main: per-node features -> 5-layer MLP via v_wmma_f32_16x16x32_f16 ->
//       weighted normalized message -> f32 global atomics into pooled[B][3]
//
// Invariant: weight tiles are zero in all padded K rows / N columns, so
// activation padding never needs masking (stale*0 == 0), and WMMA outputs
// in padded columns are exactly 0, so stores are unconditional.
// ---------------------------------------------------------------------------
__global__ void __launch_bounds__(256) poi_mlp_kernel(
    const float* __restrict__ t,      const float* __restrict__ pos,
    const float* __restrict__ poi_t,  const float* __restrict__ poi_pos,
    const int*   __restrict__ batch,
    const float* __restrict__ W1, const float* __restrict__ b1,
    const float* __restrict__ W2, const float* __restrict__ b2,
    const float* __restrict__ W3, const float* __restrict__ b3,
    const float* __restrict__ W4, const float* __restrict__ b4,
    const float* __restrict__ W5, const float* __restrict__ b5,
    float* __restrict__ pooled, int N)
{
  // f16 weights staged as [N][K=32] rows (zero padded) -> contiguous K reads.
  __shared__ _Float16 wl1[16][32];   // layer1: 2 -> 10
  __shared__ _Float16 wl2[32][32];   // layer2: 10 -> 20 (two N tiles)
  __shared__ _Float16 wl3[16][32];   // layer3: 20 -> 10
  __shared__ _Float16 wl4[16][32];   // layer4: 10 -> 5
  __shared__ float bias1[16], bias2[32], bias3[16], bias4[16], w5s[5], b5s;
  // per-wave activation scratch: 32 node rows x 32 K (f16), rows 64B aligned
  __shared__ _Float16 hbuf[NWAVES][32][32];

  const int tid = threadIdx.x;
  const int bd  = blockDim.x;

  // ---- one-time staging: zero everything (16B stores), then fill ----
  {
    v4i z; z[0] = 0; z[1] = 0; z[2] = 0; z[3] = 0;
    v4i* w1v = (v4i*)wl1;  // 64 x 16B
    v4i* w2v = (v4i*)wl2;  // 128
    v4i* w3v = (v4i*)wl3;  // 64
    v4i* w4v = (v4i*)wl4;  // 64
    v4i* hbv = (v4i*)hbuf; // 1024
    for (int idx = tid; idx < 64;   idx += bd) w1v[idx] = z;
    for (int idx = tid; idx < 128;  idx += bd) w2v[idx] = z;
    for (int idx = tid; idx < 64;   idx += bd) w3v[idx] = z;
    for (int idx = tid; idx < 64;   idx += bd) w4v[idx] = z;
    for (int idx = tid; idx < 1024; idx += bd) hbv[idx] = z;
  }
  __syncthreads();
  for (int idx = tid; idx < 10*2;  idx += bd) wl1[idx/2 ][idx%2 ] = (_Float16)W1[idx];
  for (int idx = tid; idx < 20*10; idx += bd) wl2[idx/10][idx%10] = (_Float16)W2[idx];
  for (int idx = tid; idx < 10*20; idx += bd) wl3[idx/20][idx%20] = (_Float16)W3[idx];
  for (int idx = tid; idx < 5*10;  idx += bd) wl4[idx/10][idx%10] = (_Float16)W4[idx];
  if (tid < 16) bias1[tid] = (tid < 10) ? b1[tid] : 0.0f;
  if (tid < 32) bias2[tid] = (tid < 20) ? b2[tid] : 0.0f;
  if (tid < 16) bias3[tid] = (tid < 10) ? b3[tid] : 0.0f;
  if (tid < 16) bias4[tid] = (tid < 5)  ? b4[tid] : 0.0f;
  if (tid < 5)  w5s[tid] = W5[tid];
  if (tid == 0) b5s = b5[0];
  __syncthreads();

  const int lane = tid & 31;
  const int wave = tid >> 5;
  _Float16 (* const hb)[32] = hbuf[wave];
  const int nl    = lane & 15;              // column N for B/C/D, row sel for A
  const int kh    = (lane < 16) ? 0 : 8;    // A-matrix K half offset
  const int koffB = (lane < 16) ? 0 : 8;    // B-matrix K half offset
  const int rb    = (lane < 16) ? 0 : 8;    // C/D row base

  auto cat = [](v8h lo, v8h hi) -> v16h {
    return __builtin_shufflevector(lo, hi, 0,1,2,3,4,5,6,7,8,9,10,11,12,13,14,15);
  };

  // B operand: lane = N column; K = {koffB..koffB+7, 16+koffB..23+koffB}
  auto makeB = [&](const _Float16 (*w)[32], int nrow) -> v16h {
    const _Float16* row = w[nrow] + koffB;
    return cat(*(const v8h*)(row), *(const v8h*)(row + 16));
  };
  const v16h B1  = makeB(wl1, nl);
  const v16h B2a = makeB(wl2, nl);
  const v16h B2b = makeB(wl2, 16 + nl);
  const v16h B3  = makeB(wl3, nl);
  const v16h B4  = makeB(wl4, nl);

  // A operand: lane = M row of tile; K = {kh..kh+7, 16+kh..23+kh}
  auto makeA = [&](int tb) -> v16h {
    const _Float16* row = hb[tb + nl];
    return cat(*(const v8h*)(row + kh), *(const v8h*)(row + 16 + kh));
  };
  // C initialized with per-column bias
  auto biasC = [&](const float* barr, int noff) -> v8f {
    float bv = barr[nl + noff];
    v8f c;
#pragma unroll
    for (int r = 0; r < 8; ++r) c[r] = bv;
    return c;
  };
  // ReLU as a single v_med3_f32 (median{v,0,+inf} == max(v,0); data NaN-free),
  // then f16 convert + transpose-store, unconditional.
  const float finf = __builtin_inff();
  auto storeH = [&](int tb, v8f c, int noff) {
    int nn = nl + noff;
#pragma unroll
    for (int r = 0; r < 8; ++r) {
      hb[tb + rb + r][nn] = (_Float16)__builtin_amdgcn_fmed3f(c[r], 0.0f, finf);
    }
  };

  const int gwave = blockIdx.x * NWAVES + wave;
  const int step  = gridDim.x * NWAVES * 32;   // total nodes per sweep

  for (int chunk = gwave * 32; chunk < N; chunk += step) {
    const int  i   = chunk + lane;
    const bool act = (i < N);
    float dpx = 0.f, dpy = 0.f, dpz = 0.f, r2 = 0.f, dt = 0.f;
    int bi = 0;
    if (act) {
      bi = batch[i];
      float d = t[i] - poi_t[bi];
      dt  = (d > 0.0f) ? 1.0f : ((d < 0.0f) ? -1.0f : 0.0f);
      const float* pp = pos + 3 * (size_t)i;
      const float* qp = poi_pos + 3 * (size_t)bi;
      dpx = pp[0] - qp[0];
      dpy = pp[1] - qp[1];
      dpz = pp[2] - qp[2];
      r2  = dpx*dpx + dpy*dpy + dpz*dpz;
    }
    // unconditional prefetch of next chunk (global_prefetch_b8);
    // speculative prefetch past the buffer is silently dropped (ISA 10.5)
    {
      int nx = i + step;
      __builtin_prefetch(t + nx, 0, 0);
      __builtin_prefetch(pos + 3 * (size_t)nx, 0, 0);
      __builtin_prefetch(batch + nx, 0, 0);
    }
    // stage input features: h[node][0]=sign(dt), h[node][1]=r2 (one b32 store)
    v2h feat; feat[0] = (_Float16)dt; feat[1] = (_Float16)r2;
    *(v2h*)(&hb[lane][0]) = feat;

    // two 16-node M tiles per wave; 5 WMMAs each
#pragma unroll
    for (int tb = 0; tb <= 16; tb += 16) {
      v16h a1 = makeA(tb);
      v8f  c1 = __builtin_amdgcn_wmma_f32_16x16x32_f16(false, a1, false, B1,
                    (short)0, biasC(bias1, 0), false, false);
      storeH(tb, c1, 0);

      v16h a2  = makeA(tb);
      v8f  c2a = __builtin_amdgcn_wmma_f32_16x16x32_f16(false, a2, false, B2a,
                    (short)0, biasC(bias2, 0), false, false);
      v8f  c2b = __builtin_amdgcn_wmma_f32_16x16x32_f16(false, a2, false, B2b,
                    (short)0, biasC(bias2, 16), false, false);
      storeH(tb, c2a, 0);
      storeH(tb, c2b, 16);

      v16h a3 = makeA(tb);
      v8f  c3 = __builtin_amdgcn_wmma_f32_16x16x32_f16(false, a3, false, B3,
                    (short)0, biasC(bias3, 0), false, false);
      storeH(tb, c3, 0);

      v16h a4 = makeA(tb);
      v8f  c4 = __builtin_amdgcn_wmma_f32_16x16x32_f16(false, a4, false, B4,
                    (short)0, biasC(bias4, 0), false, false);
      storeH(tb, c4, 0);
    }

    // final layer (5 -> 1) per lane + weighted normalized message + atomics
    if (act) {
      v8h hv = *(const v8h*)(hb[lane]);      // one ds_load_b128, cols 0..7
      float w = b5s;
#pragma unroll
      for (int k = 0; k < 5; ++k) w += (float)hv[k] * w5s[k];
      float inv = w / fmaxf(sqrtf(r2), EPSF);
      unsafeAtomicAdd(&pooled[3*bi+0], inv * dpx);
      unsafeAtomicAdd(&pooled[3*bi+1], inv * dpy);
      unsafeAtomicAdd(&pooled[3*bi+2], inv * dpz);
    }
  }
}

// ---------------------------------------------------------------------------
// row-wise L2 normalize of pooled -> out
// ---------------------------------------------------------------------------
__global__ void finalize_kernel(const float* __restrict__ pooled,
                                float* __restrict__ out, int Bn) {
  int i = blockIdx.x * blockDim.x + threadIdx.x;
  if (i < Bn) {
    float x = pooled[3*i+0], y = pooled[3*i+1], z = pooled[3*i+2];
    float inv = 1.0f / fmaxf(sqrtf(x*x + y*y + z*z), EPSF);
    out[3*i+0] = x * inv;
    out[3*i+1] = y * inv;
    out[3*i+2] = z * inv;
  }
}

extern "C" void kernel_launch(void* const* d_in, const int* in_sizes, int n_in,
                              void* d_out, int out_size, void* d_ws, size_t ws_size,
                              hipStream_t stream) {
  const float* t       = (const float*)d_in[0];
  const float* pos     = (const float*)d_in[1];
  const float* poi_t   = (const float*)d_in[2];
  const float* poi_pos = (const float*)d_in[3];
  const int*   batch   = (const int*)d_in[4];
  const float* W1 = (const float*)d_in[5];  const float* b1 = (const float*)d_in[6];
  const float* W2 = (const float*)d_in[7];  const float* b2 = (const float*)d_in[8];
  const float* W3 = (const float*)d_in[9];  const float* b3 = (const float*)d_in[10];
  const float* W4 = (const float*)d_in[11]; const float* b4 = (const float*)d_in[12];
  const float* W5 = (const float*)d_in[13]; const float* b5 = (const float*)d_in[14];

  const int N  = in_sizes[0];
  const int Bn = in_sizes[2];
  float* pooled = (float*)d_ws;

  zero_kernel<<<256, 256, 0, stream>>>(pooled, 3 * Bn);

  int chunks = (N + 31) / 32;                 // 32 nodes per wave-chunk
  int blocks = (chunks + NWAVES - 1) / NWAVES;
  if (blocks > 2048) blocks = 2048;
  if (blocks < 1)    blocks = 1;
  poi_mlp_kernel<<<blocks, 256, 0, stream>>>(
      t, pos, poi_t, poi_pos, batch,
      W1, b1, W2, b2, W3, b3, W4, b4, W5, b5,
      pooled, N);

  finalize_kernel<<<(Bn + 255) / 256, 256, 0, stream>>>(pooled, (float*)d_out, Bn);
}
